// Codebook_71090298684064
// MI455X (gfx1250) — matrix operand order repaired and checked by reading, not compiled, not run on hardware
//
#include <hip/hip_runtime.h>
#include <hip/hip_bf16.h>

typedef __attribute__((ext_vector_type(16))) _Float16 v16h;
typedef __attribute__((ext_vector_type(8)))  float    v8f;
typedef __attribute__((ext_vector_type(4)))  int      v4i;

#define B_ 8192
#define K_ 8
#define N_ 1024
#define D_ 256
#define CHUNKS  (D_ / 32)        // 8
#define ROWS_PER_BLK 128         // 8 waves x 16 rows
#define BLKS_PER_K (B_ / ROWS_PER_BLK)   // 64
#define NPART (K_ * BLKS_PER_K)  // 512 partial sums
#define LDS_STRIDE 272           // 256 halves + 16 pad (32B) -> conflict-free b128 reads
#define STAGE_COLS 64            // columns staged per double-buffer slot
#define NSTAGES (N_ / STAGE_COLS)        // 16
#define OPS_PER_STAGE 8          // 16B async copies per thread per stage

#if __has_builtin(__builtin_amdgcn_global_load_async_to_lds_b128)
#define ASYNC_COPY 1
#else
#define ASYNC_COPY 0
#endif

#if ASYNC_COPY
#if __has_builtin(__builtin_amdgcn_s_wait_asynccnt)
#define WAIT_ASYNC(n) __builtin_amdgcn_s_wait_asynccnt(n)
#else
#define WAIT_ASYNC(n) asm volatile("s_wait_asynccnt %0" ::"i"(n) : "memory")
#endif
#else
#define WAIT_ASYNC(n) ((void)0)
#endif

// ---------------------------------------------------------------------------
// Prep: entries f32 -> f16 (eh), plus ||e||^2 per entry row (e2).
// ---------------------------------------------------------------------------
__global__ __launch_bounds__(32) void vq_prep(const float* __restrict__ entries,
                                              _Float16* __restrict__ eh,
                                              float* __restrict__ e2) {
    const int row  = blockIdx.x;           // 0 .. K*N-1
    const int lane = threadIdx.x;
    const float* p = entries + (size_t)row * D_;
    _Float16*    q = eh      + (size_t)row * D_;
    float s = 0.f;
    #pragma unroll
    for (int j = lane; j < D_; j += 32) {
        float v = p[j];
        q[j] = (_Float16)v;                // RTNE
        s += v * v;
    }
    #pragma unroll
    for (int m = 16; m >= 1; m >>= 1) s += __shfl_xor(s, m, 32);
    if (lane == 0) e2[row] = s;
}

// ---------------------------------------------------------------------------
// Main: 256-thread block (8 waves), one k, 128 rows. 64-column codebook
// stages double-buffered in LDS via async global->LDS; per stage each wave
// runs sub-tile PAIRS with software-pipelined B-fragments so ds_load latency
// hides behind two independent WMMA chains.
// ---------------------------------------------------------------------------
__global__ __launch_bounds__(256) void vq_main(const float* __restrict__ x,
                                               const float* __restrict__ entries,
                                               const _Float16* __restrict__ eh,
                                               const float* __restrict__ e2,
                                               float* __restrict__ xq,
                                               float* __restrict__ idx_out,
                                               float* __restrict__ partials) {
    __shared__ _Float16 sbuf[2][STAGE_COLS * LDS_STRIDE];  // 2 x 34 KB
    __shared__ float    s_e2[N_];                          // 4 KB
    __shared__ int      s_idx[8][16];
    __shared__ float    s_part[8];

    const int tid  = threadIdx.x;
    const int wave = tid >> 5;
    const int lane = tid & 31;
    const int hf   = lane >> 4;
    const int lc   = lane & 15;
    const int k    = blockIdx.x / BLKS_PER_K;
    const int b0   = (blockIdx.x % BLKS_PER_K) * ROWS_PER_BLK + wave * 16;

    const _Float16* ehk = eh + (size_t)k * N_ * D_;

    // Stage ||e||^2 for this k (visibility covered by first loop barrier).
    {
        const float4* src = (const float4*)(e2 + (size_t)k * N_);
        ((float4*)s_e2)[tid] = src[tid];           // 256 threads x 16B = 4KB
    }

    // ---- Resident A fragments (16-bit A 16x32 layout), rows b0..b0+15 ----
    const float* xrow = x + ((size_t)(b0 + lc) * K_ + k) * D_;
    v16h a[CHUNKS];
    #pragma unroll
    for (int c = 0; c < CHUNKS; ++c) {
        const float* p = xrow + 32 * c + 16 * hf;
        float4 f0 = *(const float4*)(p);
        float4 f1 = *(const float4*)(p + 4);
        float4 f2 = *(const float4*)(p + 8);
        float4 f3 = *(const float4*)(p + 12);
        a[c][0]  = (_Float16)f0.x; a[c][1]  = (_Float16)f0.y;
        a[c][2]  = (_Float16)f0.z; a[c][3]  = (_Float16)f0.w;
        a[c][4]  = (_Float16)f1.x; a[c][5]  = (_Float16)f1.y;
        a[c][6]  = (_Float16)f1.z; a[c][7]  = (_Float16)f1.w;
        a[c][8]  = (_Float16)f2.x; a[c][9]  = (_Float16)f2.y;
        a[c][10] = (_Float16)f2.z; a[c][11] = (_Float16)f2.w;
        a[c][12] = (_Float16)f3.x; a[c][13] = (_Float16)f3.y;
        a[c][14] = (_Float16)f3.z; a[c][15] = (_Float16)f3.w;
    }

    // ---- Stage: 64 rows x 256 halves (32KB, contiguous in eh) -> LDS ----
    auto issue_stage = [&](int s, int buf) {
        const _Float16* gsrc = ehk + (size_t)(s * STAGE_COLS) * D_;
        #pragma unroll
        for (int j = 0; j < OPS_PER_STAGE; ++j) {
            const int o   = tid + 256 * j;        // 16B transfer index, 0..2047
            const int row = o >> 5;               // 32 x 16B per row
            _Float16* dst = &sbuf[buf][row * LDS_STRIDE + (o & 31) * 8];
#if ASYNC_COPY
            __builtin_amdgcn_global_load_async_to_lds_b128(
                (__attribute__((address_space(1))) v4i*)(gsrc + (size_t)o * 8),
                (__attribute__((address_space(3))) v4i*)dst, 0, 0);
#else
            *(uint4*)dst = *(const uint4*)(gsrc + (size_t)o * 8);
#endif
        }
    };

    float bs[8];
    int   bi[8];
    #pragma unroll
    for (int r = 0; r < 8; ++r) { bs[r] = 3.4e38f; bi[r] = 0; }

    issue_stage(0, 0);

    for (int s = 0; s < NSTAGES; ++s) {
        if (s < NSTAGES - 1) {
            issue_stage(s + 1, (s + 1) & 1);
            WAIT_ASYNC(OPS_PER_STAGE);     // stage s complete; s+1 in flight
        } else {
            WAIT_ASYNC(0);
        }
        __syncthreads();                   // stage s visible to all waves

        // Two sub-tile pairs; each pair = 2 independent accumulator chains
        // with software-pipelined B-fragment loads.
        #pragma unroll
        for (int sp = 0; sp < 2; ++sp) {
            const int sub0 = 2 * sp;
            const int n0   = s * STAGE_COLS + sub0 * 16;       // pair's first col
            const _Float16* erow0 =
                &sbuf[s & 1][(sub0 * 16 + lc) * LDS_STRIDE + 16 * hf];
            const _Float16* erow1 = erow0 + 16 * LDS_STRIDE;

            v8f acc0 = {0.f, 0.f, 0.f, 0.f, 0.f, 0.f, 0.f, 0.f};
            v8f acc1 = {0.f, 0.f, 0.f, 0.f, 0.f, 0.f, 0.f, 0.f};
            v16h bf0 = *(const v16h*)(erow0);
            v16h bf1 = *(const v16h*)(erow1);
            #pragma unroll
            for (int c = 0; c < CHUNKS; ++c) {
                v16h c0 = bf0, c1 = bf1;
                if (c < CHUNKS - 1) {                  // prefetch next chunk
                    bf0 = *(const v16h*)(erow0 + 32 * (c + 1));
                    bf1 = *(const v16h*)(erow1 + 32 * (c + 1));
                }
                acc0 = __builtin_amdgcn_wmma_f32_16x16x32_f16(
                    false, a[c], false, c0, (short)0, acc0, false, false);
                acc1 = __builtin_amdgcn_wmma_f32_16x16x32_f16(
                    false, a[c], false, c1, (short)0, acc1, false, false);
            }
            const float e2v0 = s_e2[n0 + lc];
            const float e2v1 = s_e2[n0 + 16 + lc];
            #pragma unroll
            for (int r = 0; r < 8; ++r) {
                float sc0 = e2v0 - 2.0f * acc0[r];             // argmin scores
                float sc1 = e2v1 - 2.0f * acc1[r];
                if (sc0 < bs[r]) { bs[r] = sc0; bi[r] = n0 + lc; }
                if (sc1 < bs[r]) { bs[r] = sc1; bi[r] = n0 + 16 + lc; }
            }
        }
        __syncthreads();                   // all waves done reading buf[s&1]
    }

    // Cross-lane argmin within each 16-lane group; tie -> lower index.
    #pragma unroll
    for (int r = 0; r < 8; ++r) {
        #pragma unroll
        for (int m = 8; m >= 1; m >>= 1) {
            float os = __shfl_xor(bs[r], m, 16);
            int   oi = __shfl_xor(bi[r], m, 16);
            if (os < bs[r] || (os == bs[r] && oi < bi[r])) { bs[r] = os; bi[r] = oi; }
        }
        if (lc == r) s_idx[wave][hf * 8 + r] = bi[r];     // row m = r + 8*hf
    }
    __syncthreads();

    // Gather winners (f32), write x_q, exact f32 min-distance partials.
    const float4* x4 = (const float4*)x;
    const float4* e4 = (const float4*)entries;
    float4*       q4 = (float4*)xq;
    float lsum = 0.f;
    for (int t = lane; t < 16 * (D_ / 4); t += 32) {
        const int m = t >> 6;
        const int j = t & 63;
        const int widx = s_idx[wave][m];
        float4 q  = e4[((size_t)k * N_ + widx) * (D_ / 4) + j];
        size_t xo = ((size_t)(b0 + m) * K_ + k) * (D_ / 4) + j;
        float4 xv = x4[xo];
        q4[xo] = q;
        float d0 = xv.x - q.x, d1 = xv.y - q.y, d2 = xv.z - q.z, d3 = xv.w - q.w;
        lsum += d0 * d0 + d1 * d1 + d2 * d2 + d3 * d3;
    }
    #pragma unroll
    for (int m = 16; m >= 1; m >>= 1) lsum += __shfl_xor(lsum, m, 32);
    if (lane == 0) s_part[wave] = lsum;
    if (lane < 16) idx_out[(size_t)(b0 + lane) * K_ + k] = (float)s_idx[wave][lane];
    __syncthreads();
    if (tid == 0) {
        float tot = 0.f;
        #pragma unroll
        for (int w = 0; w < 8; ++w) tot += s_part[w];     // fixed order
        partials[blockIdx.x] = tot;
    }
}

// ---------------------------------------------------------------------------
// Deterministic final reduction of 512 partials -> two loss scalars.
// ---------------------------------------------------------------------------
__global__ __launch_bounds__(256) void vq_loss(const float* __restrict__ partials,
                                               float* __restrict__ out) {
    __shared__ float s[256];
    const int tid = threadIdx.x;
    float v = 0.f;
    for (int i = tid; i < NPART; i += 256) v += partials[i];
    s[tid] = v;
    __syncthreads();
    for (int off = 128; off > 0; off >>= 1) {
        if (tid < off) s[tid] += s[tid + off];
        __syncthreads();
    }
    if (tid == 0) {
        float mean = s[0] * (1.0f / (float)(B_ * K_));
        out[0] = mean;          // dictionary_loss
        out[1] = 0.25f * mean;  // commitment_loss (BETA = 0.25)
    }
}

extern "C" void kernel_launch(void* const* d_in, const int* in_sizes, int n_in,
                              void* d_out, int out_size, void* d_ws, size_t ws_size,
                              hipStream_t stream) {
    (void)in_sizes; (void)n_in; (void)out_size; (void)ws_size;
    const float* x       = (const float*)d_in[0];   // [B,K,D] f32
    const float* entries = (const float*)d_in[1];   // [K,N,D] f32

    float* out     = (float*)d_out;
    float* xq      = out;                               // [B,K,D]
    float* idx_out = out + (size_t)B_ * K_ * D_;        // [B,K]
    float* losses  = idx_out + (size_t)B_ * K_;         // [2]

    char*     ws       = (char*)d_ws;
    _Float16* eh       = (_Float16*)ws;                                   // K*N*D f16
    float*    e2       = (float*)(ws + (size_t)K_ * N_ * D_ * sizeof(_Float16));
    float*    partials = e2 + (size_t)K_ * N_;                            // 512 f32

    vq_prep<<<K_ * N_, 32, 0, stream>>>(entries, eh, e2);
    vq_main<<<K_ * BLKS_PER_K, 256, 0, stream>>>(x, entries, eh, e2, xq, idx_out, partials);
    vq_loss<<<1, 256, 0, stream>>>(partials, losses);
}
